// KnowledgeGraphRGCN_85495618995036
// MI455X (gfx1250) — matrix operand order repaired and checked by reading, not compiled, not run on hardware
//
#include <hip/hip_runtime.h>
#include <hip/hip_bf16.h>

// RGCN 2-layer forward for MI455X (gfx1250), wave32 + WMMA bf16 + TDM.
//
// Strategy (roofline-driven):
//  - Layer 1: transform-then-aggregate. Y_r = x @ W1_r (WMMA bf16, stored bf16),
//    then per-edge scatter of 256 values (not 512) with f32 atomics into h.
//  - Layer 2: aggregate-then-transform. agg2[r,dst] += h[src]*inv (f32 atomics,
//    256 values/edge), then one WMMA GEMM sums 8 relation terms + root term.
//  - GEMMs: v_wmma_f32_16x16x32_bf16, 128x64 block tile, 8 waves.
//  - Layer-1 GEMM tiles are fetched by the Tensor Data Mover (tensor_load_to_lds)
//    with double-buffered LDS and s_wait_tensorcnt; hardware zero-fills OOB rows
//    and applies the LDS pad (40-elem row pitch) via the descriptor pad feature.

#define RREL 8
#define DIN  512
#define DHID 256
#define DOUT 512

typedef __attribute__((ext_vector_type(16))) __bf16 v16bf;
typedef __attribute__((ext_vector_type(8)))  float  v8f;

union FragU { uint4 u[2]; v16bf v; };

__device__ __forceinline__ v8f wmma_bf16(v16bf a, v16bf b, v8f c) {
  // (neg_a, A, neg_b, B, c_mod, C, reuse_a, reuse_b)
  return __builtin_amdgcn_wmma_f32_16x16x32_bf16(false, a, false, b, (short)0, c, false, false);
}

// A fragment (16x32 bf16): lane holds row m=(lane&15); K = h*16 .. h*16+15 contiguous, h=lane>>4
__device__ __forceinline__ v16bf load_fragA(const __hip_bfloat16* rowp, int half) {
  FragU f;
  f.u[0] = *(const uint4*)(rowp + half * 16);
  f.u[1] = *(const uint4*)(rowp + half * 16 + 8);
  return f.v;
}
// B fragment (32x16 bf16) from an N-major (transposed) tile: lane holds col n=(lane&15),
// K = (lane>>4)*16 .. +15 contiguous -> one 32B LDS read.
__device__ __forceinline__ v16bf load_fragB(const __hip_bfloat16* p) {
  FragU f;
  f.u[0] = *(const uint4*)(p);
  f.u[1] = *(const uint4*)(p + 8);
  return f.v;
}

// ------------------------- Tensor Data Mover support -------------------------
#if defined(__AMDGCN__) && __has_builtin(__builtin_amdgcn_tensor_load_to_lds)
#define RGCN_USE_TDM 1
#else
#define RGCN_USE_TDM 0
#endif

#if RGCN_USE_TDM
typedef __attribute__((ext_vector_type(4))) unsigned int tdm_u32x4;
typedef __attribute__((ext_vector_type(4))) int          tdm_i32x4;
typedef __attribute__((ext_vector_type(8))) int          tdm_i32x8;

// Flat addresses into LDS carry the LDS byte offset in addr[31:0] (ISA 10.2).
__device__ __forceinline__ unsigned lds_byte_off(const void* p) {
  return (unsigned)(unsigned long long)p;
}

// 2D bf16 tile load: tile is 32 elems wide x tileRows tall, from a row-major
// matrix with rowStrideElems pitch. tensor_dim1 = tensorRows so rows past the
// matrix end are hardware zero-filled. LDS pad: 4 DWORDs after every 16 DWORDs
// stored -> 80-byte (40-element) LDS row pitch, matching the fragment layout.
__device__ __forceinline__ void tdm_load_tile_bf16(
    unsigned ldsOff, const __hip_bfloat16* gtile,
    unsigned tensorRows, unsigned tileRows, unsigned rowStrideElems)
{
  const unsigned long long ga = (unsigned long long)gtile;
  tdm_u32x4 g0;
  g0.x = 1u;                                  // count=1, user descriptor
  g0.y = ldsOff;                              // lds_addr
  g0.z = (unsigned)ga;                        // global_addr[31:0]
  g0.w = (unsigned)(ga >> 32) | (2u << 30);   // global_addr[56:32] | type=2
  tdm_i32x8 g1;
  g1[0] = (1 << 16)        // data_size = 2 bytes
        | (1 << 20)        // pad_enable
        | (3 << 22)        // pad_interval: 16 DWORDs
        | (3 << 25);       // pad_amount: 4 DWORDs
  g1[1] = (int)(32u << 16);                               // tensor_dim0[15:0] = 32
  g1[2] = (int)(tensorRows << 16);                        // tensor_dim1[15:0]
  g1[3] = (int)((tensorRows >> 16) | (32u << 16));        // tensor_dim1[31:16] | tile_dim0=32
  g1[4] = (int)(tileRows & 0xffffu);                      // tile_dim1 (tile_dim2=0)
  g1[5] = (int)rowStrideElems;                            // tensor_dim0_stride[31:0]
  g1[6] = 0;                                              // stride0 hi | stride1 lo
  g1[7] = 0;
  const tdm_i32x4 z4 = {0, 0, 0, 0};
#if __clang_major__ >= 23
  const tdm_i32x8 z8 = {0, 0, 0, 0, 0, 0, 0, 0};
  __builtin_amdgcn_tensor_load_to_lds(g0, g1, z4, z4, z8, 0);
#else
  __builtin_amdgcn_tensor_load_to_lds(g0, g1, z4, z4, 0);
#endif
}
#endif  // RGCN_USE_TDM

// ---------------- generic bf16 GEMM:  C[z] = A[z] @ Bt[z]^T (+bias) ----------------
// A: [M x K] bf16 row-major. Bt: [Nn x K] bf16 (i.e. B transposed). C: f32 or bf16.
__global__ __launch_bounds__(256)
void gemm_bf16_kernel(const __hip_bfloat16* __restrict__ A,  long zA,
                      const __hip_bfloat16* __restrict__ Bt, long zB,
                      float* __restrict__ Cf, __hip_bfloat16* __restrict__ Cb, long zC,
                      const float* __restrict__ bias, int M, int Nn, int K)
{
  A  += (long)blockIdx.z * zA;
  Bt += (long)blockIdx.z * zB;
  const long cOff = (long)blockIdx.z * zC;
  const int m0 = blockIdx.y * 128, n0 = blockIdx.x * 64;
  const int tid = threadIdx.x, lane = tid & 31, w = tid >> 5;
  const int wm = w >> 1, wn = w & 1;         // 4x2 wave grid, 32x32 per wave
  v8f acc[2][2] = {};

#if RGCN_USE_TDM
  __shared__ __hip_bfloat16 sA[2][128 * 40];
  __shared__ __hip_bfloat16 sB[2][64 * 40];
  const int nk = K >> 5;
  const bool issuer = (tid < 32);            // one wave drives the TDM
  const unsigned aOff[2] = { lds_byte_off(&sA[0][0]), lds_byte_off(&sA[1][0]) };
  const unsigned bOff[2] = { lds_byte_off(&sB[0][0]), lds_byte_off(&sB[1][0]) };
  const unsigned rowsA = (unsigned)(M - m0); // HW zero-fills rows past this
  if (issuer) {
    tdm_load_tile_bf16(aOff[0], A + (long)m0 * K, rowsA, 128u, (unsigned)K);
    tdm_load_tile_bf16(bOff[0], Bt + (long)n0 * K, 64u, 64u, (unsigned)K);
  }
  for (int ik = 0; ik < nk; ++ik) {
    const int cur = ik & 1;
    if (issuer) {
      if (ik + 1 < nk) {
        const int kk2 = (ik + 1) << 5;
        tdm_load_tile_bf16(aOff[cur ^ 1], A + (long)m0 * K + kk2, rowsA, 128u, (unsigned)K);
        tdm_load_tile_bf16(bOff[cur ^ 1], Bt + (long)n0 * K + kk2, 64u, 64u, (unsigned)K);
        __builtin_amdgcn_s_wait_tensorcnt(2);  // in-order: current pair has landed
      } else {
        __builtin_amdgcn_s_wait_tensorcnt(0);
      }
    }
    __syncthreads();                          // tile `cur` visible to all waves
    const __hip_bfloat16* pA = sA[cur];
    const __hip_bfloat16* pB = sB[cur];
    const int half = lane >> 4, fr = lane & 15;
    v16bf a0 = load_fragA(pA + (wm * 32 + fr) * 40, half);
    v16bf a1 = load_fragA(pA + (wm * 32 + 16 + fr) * 40, half);
    v16bf b0 = load_fragB(pB + (wn * 32 + fr) * 40 + half * 16);
    v16bf b1 = load_fragB(pB + (wn * 32 + 16 + fr) * 40 + half * 16);
    acc[0][0] = wmma_bf16(a0, b0, acc[0][0]);
    acc[0][1] = wmma_bf16(a0, b1, acc[0][1]);
    acc[1][0] = wmma_bf16(a1, b0, acc[1][0]);
    acc[1][1] = wmma_bf16(a1, b1, acc[1][1]);
    __syncthreads();                          // done reading before buffer reuse
  }
#else
  __shared__ __hip_bfloat16 sA[128 * 40];   // 128 rows x 32 K, pad to 40
  __shared__ __hip_bfloat16 sB[64 * 40];    // 64 cols  x 32 K, pad to 40
  for (int kk = 0; kk < K; kk += 32) {
    __syncthreads();
    { // A tile: 2 threads/row, 32B each
      const int tr = tid >> 1, half = tid & 1, g = m0 + tr;
      uint4 d0 = {0u,0u,0u,0u}, d1 = d0;
      if (g < M) {
        const uint4* s = (const uint4*)(A + (long)g * K + kk + half * 16);
        d0 = s[0]; d1 = s[1];
      }
      uint4* d = (uint4*)(sA + tr * 40 + half * 16);
      d[0] = d0; d[1] = d1;
    }
    { // B tile: 4 threads/row, 16B each (Nn is a multiple of 64 -> no guard)
      const int tr = tid >> 2, q = tid & 3;
      *(uint4*)(sB + tr * 40 + q * 8) = *(const uint4*)(Bt + (long)(n0 + tr) * K + kk + q * 8);
    }
    __syncthreads();
    const int half = lane >> 4, fr = lane & 15;
    v16bf a0 = load_fragA(sA + (wm * 32 + fr) * 40, half);
    v16bf a1 = load_fragA(sA + (wm * 32 + 16 + fr) * 40, half);
    v16bf b0 = load_fragB(sB + (wn * 32 + fr) * 40 + half * 16);
    v16bf b1 = load_fragB(sB + (wn * 32 + 16 + fr) * 40 + half * 16);
    acc[0][0] = wmma_bf16(a0, b0, acc[0][0]);
    acc[0][1] = wmma_bf16(a0, b1, acc[0][1]);
    acc[1][0] = wmma_bf16(a1, b0, acc[1][0]);
    acc[1][1] = wmma_bf16(a1, b1, acc[1][1]);
  }
#endif

  #pragma unroll
  for (int i = 0; i < 2; ++i)
    #pragma unroll
    for (int j = 0; j < 2; ++j) {
      const int col = n0 + wn * 32 + j * 16 + (lane & 15);
      const float bv = bias ? bias[col] : 0.0f;
      const int rbase = m0 + wm * 32 + i * 16 + (lane >> 4) * 8;
      #pragma unroll
      for (int v = 0; v < 8; ++v) {
        const int row = rbase + v;
        if (row < M) {
          const float val = acc[i][j][v] + bv;
          if (Cb) Cb[cOff + (long)row * Nn + col] = __float2bfloat16(val);
          else    Cf[(long)row * Nn + col] = val;
        }
      }
    }
}

// ---------- layer-2 GEMM: C = sum_{t<8} bf16(Agg_t) @ W2t_t^T + Hb @ root2t^T + b2 ----------
__global__ __launch_bounds__(256)
void gemm_layer2_kernel(const float* __restrict__ Agg,         // [8][M][DHID] f32
                        const __hip_bfloat16* __restrict__ Hb, // [M][DHID] bf16
                        const __hip_bfloat16* __restrict__ Bt, // [9][DOUT][DHID] bf16
                        const float* __restrict__ bias, float* __restrict__ C, int M)
{
  const int K = DHID, Nn = DOUT;
  __shared__ __hip_bfloat16 sA[128 * 40];
  __shared__ __hip_bfloat16 sB[64 * 40];
  const int m0 = blockIdx.y * 128, n0 = blockIdx.x * 64;
  const int tid = threadIdx.x, lane = tid & 31, w = tid >> 5;
  const int wm = w >> 1, wn = w & 1;
  v8f acc[2][2] = {};

  for (int t = 0; t < 9; ++t) {
    const __hip_bfloat16* B = Bt + (long)t * Nn * K;
    for (int kk = 0; kk < K; kk += 32) {
      __syncthreads();
      { // A tile (f32->bf16 convert for relation terms; bf16 copy for root term)
        const int tr = tid >> 1, half = tid & 1, g = m0 + tr;
        __hip_bfloat16* dp = sA + tr * 40 + half * 16;
        if (t < 8) {
          if (g < M) {
            const float* s = Agg + ((long)t * M + g) * K + kk + half * 16;
            #pragma unroll
            for (int q = 0; q < 16; ++q) dp[q] = __float2bfloat16(s[q]);
          } else {
            #pragma unroll
            for (int q = 0; q < 16; ++q) dp[q] = __float2bfloat16(0.0f);
          }
        } else {
          uint4 d0 = {0u,0u,0u,0u}, d1 = d0;
          if (g < M) {
            const uint4* s = (const uint4*)(Hb + (long)g * K + kk + half * 16);
            d0 = s[0]; d1 = s[1];
          }
          ((uint4*)dp)[0] = d0; ((uint4*)dp)[1] = d1;
        }
      }
      {
        const int tr = tid >> 2, q = tid & 3;
        *(uint4*)(sB + tr * 40 + q * 8) = *(const uint4*)(B + (long)(n0 + tr) * K + kk + q * 8);
      }
      __syncthreads();
      const int half = lane >> 4, fr = lane & 15;
      v16bf a0 = load_fragA(sA + (wm * 32 + fr) * 40, half);
      v16bf a1 = load_fragA(sA + (wm * 32 + 16 + fr) * 40, half);
      v16bf b0 = load_fragB(sB + (wn * 32 + fr) * 40 + half * 16);
      v16bf b1 = load_fragB(sB + (wn * 32 + 16 + fr) * 40 + half * 16);
      acc[0][0] = wmma_bf16(a0, b0, acc[0][0]);
      acc[0][1] = wmma_bf16(a0, b1, acc[0][1]);
      acc[1][0] = wmma_bf16(a1, b0, acc[1][0]);
      acc[1][1] = wmma_bf16(a1, b1, acc[1][1]);
    }
  }

  #pragma unroll
  for (int i = 0; i < 2; ++i)
    #pragma unroll
    for (int j = 0; j < 2; ++j) {
      const int col = n0 + wn * 32 + j * 16 + (lane & 15);
      const float bv = bias[col];
      const int rbase = m0 + wm * 32 + i * 16 + (lane >> 4) * 8;
      #pragma unroll
      for (int v = 0; v < 8; ++v) {
        const int row = rbase + v;
        if (row < M) C[(long)row * Nn + col] = acc[i][j][v] + bv;
      }
    }
}

// ---------------- elementwise / graph kernels ----------------
__global__ void zero_f32_kernel(float* p, long total) {
  long i = (long)blockIdx.x * blockDim.x + threadIdx.x;
  if (i < total) p[i] = 0.0f;
}
__global__ void cvt_bf16_kernel(const float* __restrict__ s, __hip_bfloat16* __restrict__ d, long total) {
  long i = (long)blockIdx.x * blockDim.x + threadIdx.x;
  if (i < total) d[i] = __float2bfloat16(s[i]);
}
__global__ void w1t_kernel(const float* __restrict__ W1, __hip_bfloat16* __restrict__ W1t) {
  long i = (long)blockIdx.x * blockDim.x + threadIdx.x;
  const long total = (long)RREL * DIN * DHID;
  if (i >= total) return;
  int r = (int)(i / (DIN * DHID)); int rem = (int)(i % (DIN * DHID));
  int ii = rem / DHID, hh = rem % DHID;
  W1t[((long)r * DHID + hh) * DIN + ii] = __float2bfloat16(W1[i]);
}
__global__ void r1t_kernel(const float* __restrict__ root1, __hip_bfloat16* __restrict__ r1t) {
  long i = (long)blockIdx.x * blockDim.x + threadIdx.x;
  const long total = (long)DIN * DHID;
  if (i >= total) return;
  int ii = (int)(i / DHID), hh = (int)(i % DHID);
  r1t[(long)hh * DIN + ii] = __float2bfloat16(root1[i]);
}
__global__ void w2t_kernel(const float* __restrict__ W2, const float* __restrict__ root2,
                           __hip_bfloat16* __restrict__ W2t) {
  long i = (long)blockIdx.x * blockDim.x + threadIdx.x;
  const long total = 9L * DHID * DOUT;
  if (i >= total) return;
  int t = (int)(i / (DHID * DOUT)); int rem = (int)(i % (DHID * DOUT));
  int hh = rem / DOUT, o = rem % DOUT;
  float v = (t < 8) ? W2[i] : root2[rem];
  W2t[((long)t * DOUT + o) * DHID + hh] = __float2bfloat16(v);
}
__global__ void count_kernel(const int* __restrict__ et, const int* __restrict__ dst,
                             float* __restrict__ cnt, int E, int n) {
  int e = blockIdx.x * blockDim.x + threadIdx.x;
  if (e < E) atomicAdd(&cnt[(long)et[e] * n + dst[e]], 1.0f);
}
__global__ void inv_kernel(float* __restrict__ c, long total) {
  long i = (long)blockIdx.x * blockDim.x + threadIdx.x;
  if (i < total) { float v = c[i]; c[i] = (v > 0.0f) ? 1.0f / v : 0.0f; }
}
// h[dst] += Y[etype][src] * inv[etype,dst]   (one wave per edge, 8 values/lane)
__global__ void scatter_bf16_kernel(const __hip_bfloat16* __restrict__ Y, const float* __restrict__ inv,
                                    const int* __restrict__ src, const int* __restrict__ dst,
                                    const int* __restrict__ et, float* __restrict__ out, int E, int n) {
  const int e = blockIdx.x * 8 + (threadIdx.x >> 5);
  const int lane = threadIdx.x & 31;
  if (e >= E) return;
  if (lane == 0 && e + 4096 < E) {      // stream the index arrays ahead
    __builtin_prefetch(src + e + 4096, 0, 1);
    __builtin_prefetch(dst + e + 4096, 0, 1);
    __builtin_prefetch(et + e + 4096, 0, 1);
  }
  const int s = src[e], d = dst[e], r = et[e];
  const float iv = inv[(long)r * n + d];
  const __hip_bfloat16* yp = Y + ((long)r * n + s) * DHID + lane * 8;
  float* op = out + (long)d * DHID + lane * 8;
  uint4 yv = *(const uint4*)yp;
  __hip_bfloat16 yt[8]; *(uint4*)yt = yv;
  #pragma unroll
  for (int j = 0; j < 8; ++j) atomicAdd(&op[j], __bfloat162float(yt[j]) * iv);
}
// agg2[etype,dst] += h[src] * inv[etype,dst]
__global__ void scatter_f32_kernel(const float* __restrict__ H, const float* __restrict__ inv,
                                   const int* __restrict__ src, const int* __restrict__ dst,
                                   const int* __restrict__ et, float* __restrict__ agg, int E, int n) {
  const int e = blockIdx.x * 8 + (threadIdx.x >> 5);
  const int lane = threadIdx.x & 31;
  if (e >= E) return;
  if (lane == 0 && e + 4096 < E) {
    __builtin_prefetch(src + e + 4096, 0, 1);
    __builtin_prefetch(dst + e + 4096, 0, 1);
    __builtin_prefetch(et + e + 4096, 0, 1);
  }
  const int s = src[e], d = dst[e], r = et[e];
  const float iv = inv[(long)r * n + d];
  const float* hp = H + (long)s * DHID + lane * 8;
  float* ap = agg + ((long)r * n + d) * DHID + lane * 8;
  float4 v0 = ((const float4*)hp)[0], v1 = ((const float4*)hp)[1];
  const float vv[8] = {v0.x, v0.y, v0.z, v0.w, v1.x, v1.y, v1.z, v1.w};
  #pragma unroll
  for (int j = 0; j < 8; ++j) atomicAdd(&ap[j], vv[j] * iv);
}
__global__ void relu_cvt_kernel(float* __restrict__ h, __hip_bfloat16* __restrict__ hb, long total) {
  long i = (long)blockIdx.x * blockDim.x + threadIdx.x;
  if (i < total) { float v = fmaxf(h[i], 0.0f); h[i] = v; hb[i] = __float2bfloat16(v); }
}
// one wave per row of 512, L2 normalize in place
__global__ void l2norm_kernel(float* __restrict__ out, int M) {
  const int row = blockIdx.x * 8 + (threadIdx.x >> 5);
  const int lane = threadIdx.x & 31;
  if (row >= M) return;
  float4* p = (float4*)(out + (long)row * DOUT);
  float4 v[4];
  float ss = 0.0f;
  #pragma unroll
  for (int j = 0; j < 4; ++j) {
    v[j] = p[lane + j * 32];
    ss += v[j].x * v[j].x + v[j].y * v[j].y + v[j].z * v[j].z + v[j].w * v[j].w;
  }
  #pragma unroll
  for (int o = 16; o > 0; o >>= 1) ss += __shfl_xor(ss, o, 32);
  const float sc = 1.0f / fmaxf(sqrtf(ss), 1e-12f);
  #pragma unroll
  for (int j = 0; j < 4; ++j) {
    float4 t = v[j]; t.x *= sc; t.y *= sc; t.z *= sc; t.w *= sc;
    p[lane + j * 32] = t;
  }
}

static inline unsigned cdiv(long a, int b) { return (unsigned)((a + b - 1) / b); }

extern "C" void kernel_launch(void* const* d_in, const int* in_sizes, int n_in,
                              void* d_out, int out_size, void* d_ws, size_t ws_size,
                              hipStream_t stream) {
  const float* x     = (const float*)d_in[0];
  const float* W1    = (const float*)d_in[1];
  const float* root1 = (const float*)d_in[2];
  const float* b1    = (const float*)d_in[3];
  const float* W2    = (const float*)d_in[4];
  const float* root2 = (const float*)d_in[5];
  const float* b2    = (const float*)d_in[6];
  const int* src = (const int*)d_in[7];
  const int* dst = (const int*)d_in[8];
  const int* et  = (const int*)d_in[9];
  float* out = (float*)d_out;
  const int n = in_sizes[0] / DIN;   // 50000
  const int E = in_sizes[7];         // 800000
  (void)n_in; (void)out_size; (void)ws_size;

  // ---- workspace carve (256B aligned). Y1 (bf16, 205MB) aliases agg2 (f32, 410MB):
  // Y1 is dead before agg2 is zeroed (stream-ordered), so the alias is safe.
  char* base = (char*)d_ws;
  size_t off = 0;
  auto carve = [&](size_t bytes) { void* r = base + off; off = (off + bytes + 255) & ~(size_t)255; return r; };
  float*          inv  = (float*)         carve((size_t)RREL * n * 4);
  __hip_bfloat16* xb   = (__hip_bfloat16*)carve((size_t)n * DIN * 2);
  __hip_bfloat16* W1t  = (__hip_bfloat16*)carve((size_t)RREL * DHID * DIN * 2);
  __hip_bfloat16* r1t  = (__hip_bfloat16*)carve((size_t)DHID * DIN * 2);
  __hip_bfloat16* W2t  = (__hip_bfloat16*)carve(9UL * DOUT * DHID * 2);
  float*          h    = (float*)         carve((size_t)n * DHID * 4);
  __hip_bfloat16* hb   = (__hip_bfloat16*)carve((size_t)n * DHID * 2);
  float*          agg2 = (float*)         carve((size_t)RREL * n * DHID * 4);
  __hip_bfloat16* Y1   = (__hip_bfloat16*)agg2;   // aliased lifetime

  const dim3 blk(256);
  const unsigned mt = cdiv(n, 128);
  const long invTot = (long)RREL * n;

  // 0) counts buffer -> 0, then count + invert
  zero_f32_kernel<<<cdiv(invTot, 256), blk, 0, stream>>>(inv, invTot);
  count_kernel<<<cdiv(E, 256), blk, 0, stream>>>(et, dst, inv, E, n);
  inv_kernel<<<cdiv(invTot, 256), blk, 0, stream>>>(inv, invTot);

  // 1) bf16 prep: x copy + transposed weights
  cvt_bf16_kernel<<<cdiv((long)n * DIN, 256), blk, 0, stream>>>(x, xb, (long)n * DIN);
  w1t_kernel<<<cdiv((long)RREL * DIN * DHID, 256), blk, 0, stream>>>(W1, W1t);
  r1t_kernel<<<cdiv((long)DIN * DHID, 256), blk, 0, stream>>>(root1, r1t);
  w2t_kernel<<<cdiv(9L * DHID * DOUT, 256), blk, 0, stream>>>(W2, root2, W2t);

  // 2) layer 1: Y_r = x @ W1_r (bf16 out, z=relation);  h = x @ root1 + b1 (f32)
  gemm_bf16_kernel<<<dim3(DHID / 64, mt, RREL), blk, 0, stream>>>(
      xb, 0L, W1t, (long)DHID * DIN, nullptr, Y1, (long)n * DHID, nullptr, n, DHID, DIN);
  gemm_bf16_kernel<<<dim3(DHID / 64, mt, 1), blk, 0, stream>>>(
      xb, 0L, r1t, 0L, h, nullptr, 0L, b1, n, DHID, DIN);

  // 3) edge scatter into h, then ReLU (+ bf16 copy for the layer-2 root term)
  scatter_bf16_kernel<<<cdiv(E, 8), blk, 0, stream>>>(Y1, inv, src, dst, et, h, E, n);
  relu_cvt_kernel<<<cdiv((long)n * DHID, 256), blk, 0, stream>>>(h, hb, (long)n * DHID);

  // 4) layer 2 aggregation: zero agg2 (kills Y1 alias), scatter h
  zero_f32_kernel<<<cdiv((long)RREL * n * DHID, 256), blk, 0, stream>>>(agg2, (long)RREL * n * DHID);
  scatter_f32_kernel<<<cdiv(E, 8), blk, 0, stream>>>(h, inv, src, dst, et, agg2, E, n);

  // 5) layer-2 GEMM (8 relation terms + root term + b2) straight into d_out, then L2 norm
  gemm_layer2_kernel<<<dim3(DOUT / 64, mt, 1), blk, 0, stream>>>(agg2, hb, W2t, b2, out, n);
  l2norm_kernel<<<cdiv(n, 8), blk, 0, stream>>>(out, n);
}
